// Self_Attention_90993177133804
// MI455X (gfx1250) — matrix operand order repaired and checked by reading, not compiled, not run on hardware
//
#include <hip/hip_runtime.h>

// ---------------------------------------------------------------------------
// CDNA5 (gfx1250) self-attention: all matmuls via v_wmma_f32_16x16x32_bf16.
// wave32; A/B fragments use contiguous k order (k = 16*half + s); C/D as in the ISA (CDNA5 ISA).
// ---------------------------------------------------------------------------

typedef __attribute__((ext_vector_type(16))) __bf16 bf16x16;
typedef __attribute__((ext_vector_type(8)))  __bf16 bf16x8;
typedef __attribute__((ext_vector_type(8)))  float  floatx8;

union AFragU { bf16x16 v; bf16x8 h[2]; };

// A-fragment (16x32 bf16, M x K). Layout: lanes 0-15 -> K {k0..k0+15};
// lanes 16-31 -> K {k0+16..k0+31} (matches load_b_frag).
static __device__ inline bf16x16 load_a_frag(const __bf16* base, int ld, int row0, int k0) {
  const int lane = threadIdx.x & 31;
  const int m = lane & 15, hf = lane >> 4;
  AFragU f;
  const __bf16* p = base + (size_t)(row0 + m) * ld + k0 + 16 * hf;
  f.h[0] = *(const bf16x8*)(p);
  f.h[1] = *(const bf16x8*)(p + 8);
  return f.v;
}

// B-fragment (32x16 bf16, K x N), read from a row-major [N, K] matrix
// (so B[k][n] = M[n0+n][k0+k]).  Contiguous k order (as A): lanes 0-15 hold K=k0..k0+15,
// lanes 16-31 hold K=k0+16..k0+31 -> one contiguous 32B load per lane.
static __device__ inline bf16x16 load_b_frag(const __bf16* base, int ld, int n0, int k0) {
  const int lane = threadIdx.x & 31;
  const int n = lane & 15, hf = lane >> 4;
  const __bf16* p = base + (size_t)(n0 + n) * ld + k0 + 16 * hf;
  return *(const bf16x16*)(p);
}

static __device__ inline floatx8 wmma_bf16(bf16x16 a, bf16x16 b, floatx8 c) {
  return __builtin_amdgcn_wmma_f32_16x16x32_bf16(false, a, false, b, (short)0, c,
                                                 false, false);
}

// ---------------------------------------------------------------------------
// fp32 -> bf16 conversion (grid-stride)
// ---------------------------------------------------------------------------
__global__ void f32_to_bf16_kernel(const float* __restrict__ in,
                                   __bf16* __restrict__ out, int n) {
  int i = blockIdx.x * blockDim.x + threadIdx.x;
  int stride = gridDim.x * blockDim.x;
  for (; i < n; i += stride) out[i] = (__bf16)in[i];
}

// ---------------------------------------------------------------------------
// GEMM: out[M, Nout] = A[M, K](bf16) * W[Nout, K]^T(bf16) (+ bias), fp32 out.
// Block tile 256x128, K-chunk 32.  8 waves in a 4x2 grid; each wave owns a
// 64x64 tile -> 16 WMMAs per 8 fragment loads (2x the FLOP/LDS-byte of a
// 32x64 tile), which is what CDNA5's WMMA rate vs LDS bandwidth rewards.
// ---------------------------------------------------------------------------
__global__ __launch_bounds__(256) void gemm_bf16_wmma_kernel(
    const __bf16* __restrict__ A, const __bf16* __restrict__ W,
    const float* __restrict__ bias, float* __restrict__ out,
    int Kdim, int Nout) {
  __shared__ __attribute__((aligned(32))) __bf16 As[256 * 32];  // 16 KB
  __shared__ __attribute__((aligned(32))) __bf16 Bs[128 * 32];  //  8 KB

  const int tid = threadIdx.x;
  const int lane = tid & 31, w = tid >> 5;
  const int wm = w & 3, wn = w >> 2;           // 4x2 wave grid (64x64 tiles)
  const int m0 = blockIdx.y * 256, n0 = blockIdx.x * 128;
  const int hf = lane >> 4, ln = lane & 15;

  floatx8 zero8 = {};
  floatx8 acc[4][4];
#pragma unroll
  for (int i = 0; i < 4; ++i)
#pragma unroll
    for (int j = 0; j < 4; ++j) acc[i][j] = zero8;

  for (int kk = 0; kk < Kdim; kk += 32) {
    // Cooperative stage.  As: 256x32 bf16 = 1024 x 16B chunks (4/thread);
    //                     Bs: 128x32 bf16 =  512 x 16B chunks (2/thread).
#pragma unroll
    for (int c = 0; c < 4; ++c) {
      int ch = tid + c * 256;          // 0..1023
      int row = ch >> 2;               // 4 chunks per 32-wide row
      int ko = (ch & 3) * 8;
      *(bf16x8*)(&As[row * 32 + ko]) =
          *(const bf16x8*)(A + (size_t)(m0 + row) * Kdim + kk + ko);
    }
#pragma unroll
    for (int c = 0; c < 2; ++c) {
      int ch = tid + c * 256;          // 0..511
      int row = ch >> 2;
      int ko = (ch & 3) * 8;
      *(bf16x8*)(&Bs[row * 32 + ko]) =
          *(const bf16x8*)(W + (size_t)(n0 + row) * Kdim + kk + ko);
    }
    if (kk + 32 < Kdim) {  // speculative prefetch of next K-chunk (global_prefetch_b8)
      __builtin_prefetch(A + (size_t)(m0 + tid) * Kdim + kk + 32, 0, 1);
      __builtin_prefetch(W + (size_t)(n0 + (tid >> 1)) * Kdim + kk + 32, 0, 1);
    }
    __syncthreads();

    bf16x16 a[4], b[4];
#pragma unroll
    for (int i = 0; i < 4; ++i) a[i] = load_a_frag(As, 32, wm * 64 + 16 * i, 0);
#pragma unroll
    for (int j = 0; j < 4; ++j) b[j] = load_b_frag(Bs, 32, wn * 64 + 16 * j, 0);
#pragma unroll
    for (int i = 0; i < 4; ++i)
#pragma unroll
      for (int j = 0; j < 4; ++j) acc[i][j] = wmma_bf16(a[i], b[j], acc[i][j]);
    __syncthreads();
  }

  // Epilogue: C/D layout -> vgpr r holds (row r + 8*hf, col ln).
#pragma unroll
  for (int i = 0; i < 4; ++i)
#pragma unroll
    for (int j = 0; j < 4; ++j) {
      int colg = n0 + wn * 64 + 16 * j + ln;
      float bb = bias ? bias[colg] : 0.0f;
#pragma unroll
      for (int r = 0; r < 8; ++r) {
        int rowg = m0 + wm * 64 + 16 * i + r + 8 * hf;
        out[(size_t)rowg * Nout + colg] = acc[i][j][r] + bb;
      }
    }
}

// ---------------------------------------------------------------------------
// RoPE + head split: qkv f32 [B*N, 3072] -> q,k,v bf16 [B,H,N,64].
// HD^-0.5 folded into q.
// ---------------------------------------------------------------------------
__global__ void rope_split_kernel(const float* __restrict__ qkv,
                                  const float* __restrict__ fcos,
                                  const float* __restrict__ fsin,
                                  __bf16* __restrict__ q, __bf16* __restrict__ k,
                                  __bf16* __restrict__ v, int total) {
  int idx = blockIdx.x * blockDim.x + threadIdx.x;
  if (idx >= total) return;
  int p = idx & 31;            // rotary pair  (HD/2 = 32)
  int h = (idx >> 5) & 15;     // head         (H = 16)
  int n = (idx >> 9) & 2047;   // sequence pos (N = 2048)
  int b = idx >> 20;           // batch        (B = 4)

  size_t base = ((size_t)b * 2048 + n) * 3072;
  int col = h * 64 + 2 * p;
  float c = fcos[n * 32 + p], s = fsin[n * 32 + p];

  float qa = qkv[base + col],        qb = qkv[base + col + 1];
  float ka = qkv[base + 1024 + col], kb = qkv[base + 1024 + col + 1];
  float va = qkv[base + 2048 + col], vb = qkv[base + 2048 + col + 1];

  const float scale = 0.125f;  // 64^-0.5
  size_t o = (((size_t)(b * 16 + h)) * 2048 + n) * 64 + 2 * p;
  q[o]     = (__bf16)((qa * c - qb * s) * scale);
  q[o + 1] = (__bf16)((qa * s + qb * c) * scale);
  k[o]     = (__bf16)(ka * c - kb * s);
  k[o + 1] = (__bf16)(ka * s + kb * c);
  v[o]     = (__bf16)va;
  v[o + 1] = (__bf16)vb;
}

// ---------------------------------------------------------------------------
// Flash attention: grid (N/128, B*H), 256 threads (8 waves), each wave owns a
// 16-row Q tile.  64-key chunks staged in LDS (V transposed) -> 16 WMMAs per
// 3 barriers; online softmax with wave32 half-lane shuffles.
// ---------------------------------------------------------------------------
__global__ __launch_bounds__(256) void flash_attn_kernel(
    const __bf16* __restrict__ Qg, const __bf16* __restrict__ Kg,
    const __bf16* __restrict__ Vg, __bf16* __restrict__ ctx, int N) {
  __shared__ __attribute__((aligned(32))) __bf16 Ks[64 * 64];    // [key][d]  8 KB
  __shared__ __attribute__((aligned(32))) __bf16 Vt[64 * 64];    // [d][key]  8 KB
  __shared__ __attribute__((aligned(32))) __bf16 Ps[8][16 * 64]; // per-wave P 16 KB

  const int tid = threadIdx.x, lane = tid & 31, w = tid >> 5;
  const int hf = lane >> 4, ln = lane & 15;
  const int bh = blockIdx.y;
  const int b = bh >> 4, h = bh & 15;

  const __bf16* Qb = Qg + (size_t)bh * N * 64;
  const __bf16* Kb = Kg + (size_t)bh * N * 64;
  const __bf16* Vb = Vg + (size_t)bh * N * 64;
  const int r0 = blockIdx.x * 128 + w * 16;

  const bf16x16 aq0 = load_a_frag(Qb, 64, r0, 0);
  const bf16x16 aq1 = load_a_frag(Qb, 64, r0, 32);

  float mrow[8], lrow[8];
  floatx8 zero8 = {};
  floatx8 o[4];
#pragma unroll
  for (int r = 0; r < 8; ++r) { mrow[r] = -1e30f; lrow[r] = 0.0f; }
#pragma unroll
  for (int j = 0; j < 4; ++j) o[j] = zero8;

  for (int kc = 0; kc < N; kc += 64) {
    __syncthreads();
    // Stage 64-key chunk: K direct, V transposed.  64x64 bf16 = 512 x 16B
    // chunks per tensor -> 2 chunks/thread each.
#pragma unroll
    for (int c = 0; c < 2; ++c) {
      int ch = tid + c * 256;          // 0..511
      int row = ch >> 3;               // 8 chunks per 64-wide row
      int ko = (ch & 7) * 8;
      *(bf16x8*)(&Ks[row * 64 + ko]) =
          *(const bf16x8*)(Kb + (size_t)(kc + row) * 64 + ko);
      bf16x8 vv = *(const bf16x8*)(Vb + (size_t)(kc + row) * 64 + ko);
#pragma unroll
      for (int j = 0; j < 8; ++j) Vt[(ko + j) * 64 + row] = vv[j];
    }
    __syncthreads();

    // S = Q * K^T : four 16x16 score tiles (keys kc+16t), d contracted in 2
    // WMMAs each -> 8 WMMAs.
    floatx8 s[4];
#pragma unroll
    for (int t = 0; t < 4; ++t) {
      s[t] = zero8;
      s[t] = wmma_bf16(aq0, load_b_frag(Ks, 64, t * 16, 0), s[t]);
      s[t] = wmma_bf16(aq1, load_b_frag(Ks, 64, t * 16, 32), s[t]);
    }

    // Online softmax per row (vgpr r <-> row r + 8*hf; cols across 16 lanes).
#pragma unroll
    for (int r = 0; r < 8; ++r) {
      float mx = fmaxf(fmaxf(s[0][r], s[1][r]), fmaxf(s[2][r], s[3][r]));
      mx = fmaxf(mx, __shfl_xor(mx, 1, 32));
      mx = fmaxf(mx, __shfl_xor(mx, 2, 32));
      mx = fmaxf(mx, __shfl_xor(mx, 4, 32));
      mx = fmaxf(mx, __shfl_xor(mx, 8, 32));
      float mnew = fmaxf(mrow[r], mx);
      float alpha = __expf(mrow[r] - mnew);
      mrow[r] = mnew;
      float p0 = __expf(s[0][r] - mnew);
      float p1 = __expf(s[1][r] - mnew);
      float p2 = __expf(s[2][r] - mnew);
      float p3 = __expf(s[3][r] - mnew);
      float rs = (p0 + p1) + (p2 + p3);
      rs += __shfl_xor(rs, 1, 32);
      rs += __shfl_xor(rs, 2, 32);
      rs += __shfl_xor(rs, 4, 32);
      rs += __shfl_xor(rs, 8, 32);
      lrow[r] = lrow[r] * alpha + rs;
#pragma unroll
      for (int j = 0; j < 4; ++j) o[j][r] *= alpha;
      int prow = r + 8 * hf;                 // C-layout -> LDS [row][col]
      Ps[w][prow * 64 + ln]      = (__bf16)p0;
      Ps[w][prow * 64 + ln + 16] = (__bf16)p1;
      Ps[w][prow * 64 + ln + 32] = (__bf16)p2;
      Ps[w][prow * 64 + ln + 48] = (__bf16)p3;
    }
    __syncthreads();  // reshape P: C-layout -> A-fragment layout via LDS

    // O += P(16x64) * V(64x64): keys contracted in two 32-deep WMMA passes,
    // V columns read from the transposed LDS tile -> 8 WMMAs.
    const bf16x16 ap0 = load_a_frag(&Ps[w][0], 64, 0, 0);
    const bf16x16 ap1 = load_a_frag(&Ps[w][0], 64, 0, 32);
#pragma unroll
    for (int j = 0; j < 4; ++j) {
      o[j] = wmma_bf16(ap0, load_b_frag(Vt, 64, j * 16, 0), o[j]);
      o[j] = wmma_bf16(ap1, load_b_frag(Vt, 64, j * 16, 32), o[j]);
    }
  }

  // Finalize: O /= l, write ctx as [B, N, H, 64] bf16 (== [B,N,C] for proj).
#pragma unroll
  for (int r = 0; r < 8; ++r) {
    float inv = 1.0f / lrow[r];
    int nrow = r0 + r + 8 * hf;
    size_t base = (((size_t)b * N + nrow) * 16 + h) * 64;
#pragma unroll
    for (int j = 0; j < 4; ++j)
      ctx[base + j * 16 + ln] = (__bf16)(o[j][r] * inv);
  }
}

// ---------------------------------------------------------------------------
// Host-side orchestration
// ---------------------------------------------------------------------------
extern "C" void kernel_launch(void* const* d_in, const int* in_sizes, int n_in,
                              void* d_out, int out_size, void* d_ws, size_t ws_size,
                              hipStream_t stream) {
  (void)in_sizes; (void)n_in; (void)out_size; (void)ws_size;
  const float* x      = (const float*)d_in[0];  // [4,2048,1024]
  const float* w_qkv  = (const float*)d_in[1];  // [3072,1024]
  const float* w_proj = (const float*)d_in[2];  // [1024,1024]
  const float* b_proj = (const float*)d_in[3];  // [1024]
  const float* fcos   = (const float*)d_in[4];  // [2048,32]
  const float* fsin   = (const float*)d_in[5];  // [2048,32]
  float* out = (float*)d_out;                   // [4,2048,1024] f32

  const int M = 4 * 2048;     // 8192 rows
  const int C = 1024;
  const size_t nX = (size_t)M * C;          // 8,388,608
  const size_t nWq = (size_t)3 * C * C;     // 3,145,728
  const size_t nWp = (size_t)C * C;         // 1,048,576

  char* ws = (char*)d_ws;
  size_t off = 0;
  auto carve = [&](size_t bytes) -> char* {
    char* p = ws + off;
    off += (bytes + 255) & ~(size_t)255;
    return p;
  };
  __bf16* xbf    = (__bf16*)carve(nX * 2);        // also reused as ctx
  __bf16* wqkvbf = (__bf16*)carve(nWq * 2);
  __bf16* wprojbf= (__bf16*)carve(nWp * 2);
  float*  qkvf   = (float*)carve((size_t)M * 3 * C * 4);
  __bf16* qbf    = (__bf16*)carve(nX * 2);
  __bf16* kbf    = (__bf16*)carve(nX * 2);
  __bf16* vbf    = (__bf16*)carve(nX * 2);
  __bf16* ctxbf  = xbf;  // x is dead after QKV GEMM; reuse its slot

  // 1) converts
  f32_to_bf16_kernel<<<4096, 256, 0, stream>>>(x, xbf, (int)nX);
  f32_to_bf16_kernel<<<4096, 256, 0, stream>>>(w_qkv, wqkvbf, (int)nWq);
  f32_to_bf16_kernel<<<2048, 256, 0, stream>>>(w_proj, wprojbf, (int)nWp);

  // 2) QKV projection: [8192,1024] x [3072,1024]^T -> [8192,3072] f32
  gemm_bf16_wmma_kernel<<<dim3(3 * C / 128, M / 256), 256, 0, stream>>>(
      xbf, wqkvbf, nullptr, qkvf, C, 3 * C);

  // 3) RoPE + split to [B,H,N,64] bf16 (scale folded into q)
  const int ropeTotal = 4 * 2048 * 16 * 32;  // one thread per rotary pair
  rope_split_kernel<<<ropeTotal / 256, 256, 0, stream>>>(qkvf, fcos, fsin,
                                                         qbf, kbf, vbf, ropeTotal);

  // 4) flash attention -> ctx [B,N,C] bf16
  flash_attn_kernel<<<dim3(2048 / 128, 4 * 16), 256, 0, stream>>>(
      qbf, kbf, vbf, ctxbf, 2048);

  // 5) output projection + bias -> d_out f32
  gemm_bf16_wmma_kernel<<<dim3(C / 128, M / 256), 256, 0, stream>>>(
      ctxbf, wprojbf, b_proj, out, C, C);
}
